// RNNModel_4904852652502
// MI455X (gfx1250) — matrix operand order repaired and checked by reading, not compiled, not run on hardware
//
#include <hip/hip_runtime.h>

// ---------------------------------------------------------------------------
// DARTS RNN scan on MI455X (gfx1250): persistent per-batch-tile blocks,
// bf16 WMMA 16x16x32 GEMMs, LDS-staged recurrent states, L2-resident
// pre-packed bf16 weights.
// v3: rolled k-loop (#pragma unroll 1) to keep B fragments streaming from
// global (L2) instead of spilling to scratch; one A fragment feeds 8 WMMAs.
// ---------------------------------------------------------------------------

typedef __bf16 bf16_t;
typedef bf16_t v16bf __attribute__((ext_vector_type(16)));
typedef float  v8f   __attribute__((ext_vector_type(8)));
typedef unsigned int v4u __attribute__((ext_vector_type(4)));

union FragU { v4u u[2]; v16bf b; };   // 32 bytes: one wmma bf16 A/B fragment per lane

#define NHID    512
#define T_STEPS 400
#define BATCH   256

// packed-weight layout (ushort elements) inside d_ws:
//   W0:  2048 fragments (ntile 0..63, ktile 0..31), frag = ntile*32 + kt
//   Ws[i]: 1024 fragments (ntile 0..63, ktile 0..15), frag = ntile*16 + kt
// each fragment = 32 lanes * 16 bf16 = 512 elements (lane-major, 32B/lane)
#define W0_ELEMS   (2048u * 512u)          // 1,048,576
#define WN_ELEMS   (1024u * 512u)          // 524,288 per node

__device__ __forceinline__ unsigned short f2bf(float f) {
    unsigned u = __float_as_uint(f);
    u += 0x7FFFu + ((u >> 16) & 1u);       // round-to-nearest-even
    return (unsigned short)(u >> 16);
}
__device__ __forceinline__ float bf2f(unsigned short h) {
    return __uint_as_float(((unsigned)h) << 16);
}
// gate math on the TRANS pipe: v_exp_f32 + v_rcp_f32 (co-executes with WMMA)
__device__ __forceinline__ float sigmoidf_(float x) {
    return __builtin_amdgcn_rcpf(1.0f + __expf(-x));
}
__device__ __forceinline__ float tanhf_(float x) {
    return __builtin_fmaf(2.0f, __builtin_amdgcn_rcpf(1.0f + __expf(-2.0f * x)), -1.0f);
}

// Compute all 8 output tiles a wave owns (4 c-tiles, 4 h-tiles) in ONE k-loop.
// A fragments (LDS) are loaded once per kt and feed 8 WMMAs; the 8 B tiles are
// addressed off a single pointer with constant immediate offsets.
// A layout: lane(l<16)=row l holds K=kt*32+0..15,
// lane(l>=16)=row l-16 holds K=kt*32+16..31.
// B packed: lane<16 holds K=kt*32+0..15 of its column, lane>=16 holds +16..31.
// NOTE: k-loop deliberately NOT unrolled — full unroll makes the compiler
// hoist all B loads and spill fragments to scratch (seen in round 2).
__device__ __forceinline__ void gemm_block(
    const unsigned short* __restrict__ Abuf, int lda, int ktiles,
    const unsigned short* __restrict__ Wp, int w, int lane, v8f acc[8])
{
    const int m    = lane & 15;
    const int half = lane >> 4;
    const unsigned short* arow = Abuf + m * lda + half * 16;
    // base pointer = tile (4w) c-half, fragment kt=0, this lane
    const unsigned short* bbase =
        Wp + ((size_t)(4 * w * ktiles) * 32 + lane) * 16;
    const int tstride = ktiles * 512;            // elems between adjacent n-tiles

    #pragma unroll 1
    for (int kt = 0; kt < ktiles; ++kt) {
        FragU a;
        const unsigned short* ap = arow + kt * 32;
        a.u[0] = *(const v4u*)(ap);              // ds_load_b128
        a.u[1] = *(const v4u*)(ap + 8);
        #pragma unroll
        for (int p = 0; p < 4; ++p) {
            FragU bc;
            const unsigned short* pc = bbase + (size_t)p * tstride;          // c tile p
            bc.u[0] = *(const v4u*)(pc);         // global_load_b128 (L2)
            bc.u[1] = *(const v4u*)(pc + 8);
            acc[p] = __builtin_amdgcn_wmma_f32_16x16x32_bf16(
                         false, a.b, false, bc.b, (short)0, acc[p], false, false);
            FragU bh;
            const unsigned short* ph = bbase + (size_t)(32 + p) * tstride;   // h tile p
            bh.u[0] = *(const v4u*)(ph);
            bh.u[1] = *(const v4u*)(ph + 8);
            acc[4 + p] = __builtin_amdgcn_wmma_f32_16x16x32_bf16(
                             false, a.b, false, bh.b, (short)0, acc[4 + p], false, false);
        }
        bbase += 512;                            // next K fragment
    }
}

// s = sp + sigmoid(c) * (act(h) - sp); C/D layout: vgpr r, lane<16 -> M=r,
// lane>=16 -> M=8+r, N = lane&15.
__device__ __forceinline__ void combine_state(
    const v8f& acc_c, const v8f& acc_h,
    const unsigned short* __restrict__ pred, int ldp,
    unsigned short* __restrict__ store_buf,   // nullptr if state never read again
    int colbase, int lane, int act, float* macc)  // macc nullptr -> not in mean
{
    const int n    = lane & 15;
    const int half = lane >> 4;
    #pragma unroll
    for (int r = 0; r < 8; ++r) {
        const int mr  = r + half * 8;
        const int col = colbase + n;
        float sp = bf2f(pred[mr * ldp + col]);
        float c  = acc_c[r];
        float h  = acc_h[r];
        float av;
        if      (act == 0) av = tanhf_(h);
        else if (act == 1) av = fmaxf(h, 0.0f);
        else if (act == 2) av = sigmoidf_(h);
        else               av = h;
        float s = __builtin_fmaf(sigmoidf_(c), av - sp, sp);
        if (store_buf) store_buf[mr * 512 + col] = f2bf(s);
        if (macc) macc[r] += s;
    }
}

// --------------------------- weight pack kernel ----------------------------
__global__ __launch_bounds__(256)
void pack_weights(const float* __restrict__ W0, const float* __restrict__ Ws,
                  unsigned short* __restrict__ Wp)
{
    const int gid   = blockIdx.x * 256 + threadIdx.x;
    const int total = (2048 + 8 * 1024) * 32;
    if (gid >= total) return;
    const int frag = gid >> 5;
    const int lane = gid & 31;
    const int nlo  = lane & 15;
    const int half = lane >> 4;

    const float* src; int ntile, kt; size_t dstbase;
    if (frag < 2048) {                        // W0 [1024][1024]
        ntile = frag >> 5; kt = frag & 31;
        src = W0;
        dstbase = (size_t)frag * 512;
    } else {                                  // Ws [8][512][1024]
        const int rem  = frag - 2048;
        const int node = rem >> 10;
        const int f2   = rem & 1023;
        ntile = f2 >> 4; kt = f2 & 15;
        src = Ws + (size_t)node * 512 * 1024;
        dstbase = W0_ELEMS + (size_t)node * WN_ELEMS + (size_t)f2 * 512;
    }
    const int ncol  = ntile * 16 + nlo;
    const int kbase = kt * 32 + half * 16;
    unsigned short* d = Wp + dstbase + (size_t)lane * 16;
    #pragma unroll
    for (int tt = 0; tt < 16; ++tt)
        d[tt] = f2bf(src[(size_t)(kbase + tt) * 1024 + ncol]);
}

// ------------------------------ main kernel --------------------------------
__global__ __launch_bounds__(256, 1)
void rnn_persist(const float* __restrict__ inputs, const float* __restrict__ hidden,
                 const unsigned short* __restrict__ Wp, float* __restrict__ out)
{
    extern __shared__ unsigned short smem[];
    unsigned short* xh = smem;                 // [16][1024] bf16: x | h_prev
    unsigned short* Sb = smem + 16 * 1024;     // 5 slots of [16][512] bf16

    const int tid  = threadIdx.x;
    const int lane = tid & 31;
    const int w    = tid >> 5;                 // wave 0..7 owns cols [64w,64w+64)
    const int b0   = blockIdx.x * 16;          // batch tile

    for (int t = 0; t < T_STEPS; ++t) {
        // ---- stage x_t (f32 -> bf16) into xh[:, 0:512]; h at t==0 ----
        const float* xsrc = inputs + ((size_t)t * BATCH + b0) * NHID;
        for (int i = tid; i < 16 * NHID; i += 256) {
            const int m = i >> 9, k = i & 511;
            xh[m * 1024 + k] = f2bf(xsrc[m * NHID + k]);
        }
        if (t == 0) {
            const float* hsrc = hidden + (size_t)b0 * NHID;
            for (int i = tid; i < 16 * NHID; i += 256) {
                const int m = i >> 9, k = i & 511;
                xh[m * 1024 + 512 + k] = f2bf(hsrc[m * NHID + k]);
            }
        }
        // warm next step's x tile into cache (global_prefetch_b8)
        if (t + 1 < T_STEPS) {
            const float* xn = inputs + ((size_t)(t + 1) * BATCH + b0) * NHID;
            __builtin_prefetch(xn + tid * 32, 0, 0);   // one 128B line per thread
        }
        __syncthreads();

        float macc[4][8];
        #pragma unroll
        for (int p = 0; p < 4; ++p)
            #pragma unroll
            for (int r = 0; r < 8; ++r) macc[p][r] = 0.0f;

        // ---- s0: [16,1024] @ W0 -> gate+tanh highway against h_prev ----
        {
            v8f acc[8];
            #pragma unroll
            for (int q = 0; q < 8; ++q) acc[q] = (v8f){};
            gemm_block(xh, 1024, 32, Wp, w, lane, acc);
            #pragma unroll
            for (int p = 0; p < 4; ++p)
                combine_state(acc[p], acc[4 + p], xh + 512, 1024, Sb /*slot0*/,
                              (4 * w + p) * 16, lane, /*tanh*/0, nullptr);
        }
        __syncthreads();

        // ---- DAG nodes: state i+1 = f(state pred_i) ----
        // preds:  s0 s1 s1 s1 s2 s5 s3 s5   (slot map: s0->0 s1->1 s2->2 s3->3 s5->4)
        // stores: only states read later (s1,s2,s3,s5) go to LDS
        const int pred_slot[8] = {0, 1, 1, 1, 2, 4, 3, 4};
        const int out_slot[8]  = {1, 2, 3, -1, 4, -1, -1, -1};
        const int actc[8]      = {0, 1, 1, 3, 0, 2, 0, 1}; // tanh,relu,relu,id,tanh,sig,tanh,relu

        #pragma unroll
        for (int i = 0; i < 8; ++i) {
            const unsigned short* pb = Sb + pred_slot[i] * 8192;
            const unsigned short* Wn = Wp + W0_ELEMS + (size_t)i * WN_ELEMS;
            unsigned short* ob = (out_slot[i] >= 0) ? (Sb + out_slot[i] * 8192)
                                                    : (unsigned short*)nullptr;
            v8f acc[8];
            #pragma unroll
            for (int q = 0; q < 8; ++q) acc[q] = (v8f){};
            gemm_block(pb, 512, 16, Wn, w, lane, acc);
            #pragma unroll
            for (int p = 0; p < 4; ++p)
                combine_state(acc[p], acc[4 + p], pb, 512, ob,
                              (4 * w + p) * 16, lane, actc[i], macc[p]);
            // s1 needed by node1; s2/s3 needed by nodes 4/6; s5 by nodes 5/7
            if (i == 0 || i == 3 || i == 4) __syncthreads();
        }

        // ---- h_new = mean(s1..s8): emit f32 + feed back bf16 ----
        const int n = lane & 15, half = lane >> 4;
        #pragma unroll
        for (int p = 0; p < 4; ++p) {
            const int colb = (4 * w + p) * 16;
            #pragma unroll
            for (int r = 0; r < 8; ++r) {
                const int mr = r + half * 8;
                const float v = macc[p][r] * 0.125f;
                out[((size_t)t * BATCH + b0 + mr) * NHID + colb + n] = v;
                xh[mr * 1024 + 512 + colb + n] = f2bf(v);
                if (t == T_STEPS - 1)
                    out[(size_t)T_STEPS * BATCH * NHID +
                        (size_t)(b0 + mr) * NHID + colb + n] = v;
            }
        }
        __syncthreads();   // protect xh / state slots before next step
    }
}

// ------------------------------- launcher ----------------------------------
extern "C" void kernel_launch(void* const* d_in, const int* in_sizes, int n_in,
                              void* d_out, int out_size, void* d_ws, size_t ws_size,
                              hipStream_t stream) {
    const float* inputs = (const float*)d_in[0];   // [400,256,512] f32
    const float* hidden = (const float*)d_in[1];   // [1,256,512]   f32
    const float* W0     = (const float*)d_in[2];   // [1024,1024]   f32
    const float* Ws     = (const float*)d_in[3];   // [8,512,1024]  f32
    float* out          = (float*)d_out;           // hiddens ++ h_last
    unsigned short* Wp  = (unsigned short*)d_ws;   // 10 MB packed bf16 weights

    // one-time-per-call weight repack (deterministic, L2-resident afterwards)
    pack_weights<<<1280, 256, 0, stream>>>(W0, Ws, Wp);

    const int smem_bytes = 16 * 1024 * 2 + 5 * 16 * 512 * 2;   // 114688 B
    (void)hipFuncSetAttribute(reinterpret_cast<const void*>(rnn_persist),
                              hipFuncAttributeMaxDynamicSharedMemorySize, smem_bytes);
    rnn_persist<<<16, 256, smem_bytes, stream>>>(inputs, hidden, Wp, out);
}
